// VisionMamba_55336358642834
// MI455X (gfx1250) — matrix operand order repaired and checked by reading, not compiled, not run on hardware
//
#include <hip/hip_runtime.h>
#include <hip/hip_bf16.h>

typedef __attribute__((ext_vector_type(16))) __bf16 v16bf;
typedef __attribute__((ext_vector_type(8)))  __bf16 v8bf;
typedef __attribute__((ext_vector_type(8)))  float  v8f;

#define EMBED  768
#define DEPTH  12
#define NSTATE 16
#define DTRK   48
#define KCONV  4
#define NH     12
#define DKH    64
#define MLPD   3072
#define NCLS   1000
#define PS     16
#define IMG    224
#define NPATCH 196
#define BATCH  16
#define SEQ    197
#define MTOK   (BATCH*SEQ)   /* 3152, multiple of 16 */
#define LP     224           /* padded attention length, multiple of 32 */
#define NKP    256           /* K-pack rows (N side of scores GEMM), multiple of 64 */
#define BHH    (BATCH*NH)    /* 192 batched heads */

__device__ __forceinline__ float gelu_f(float x){ return 0.5f*x*(1.f+erff(x*0.70710678118654752f)); }
__device__ __forceinline__ float softplus_f(float x){ return (x>20.f)? x : log1pf(__expf(x)); }

// ---------------------------------------------------------------------------
// GEMM-NT (bf16 x bf16 -> f32 acc) via v_wmma_f32_16x16x32_bf16.
// C[m,n] = sum_k A[m,k]*B[n,k] (+bias[n]) (act) (+resid[m,n])
// One wave computes a 16(M) x 64(N) strip: A fragment reused over 4 B tiles.
// REQUIREMENT: Nalloc (allocated rows of B) is a multiple of 64, so the K-loop
// is completely branch-free; partial N is masked only at the store.
// ---------------------------------------------------------------------------
__global__ __launch_bounds__(256)
void gemm_nt_bf16(const __bf16* __restrict__ A, const __bf16* __restrict__ B,
                  float* __restrict__ Cf, __bf16* __restrict__ Ch,
                  const float* __restrict__ bias, const float* __restrict__ resid,
                  int Mtiles, int K,
                  int lda, int ldb, int ldcf, int ldch,
                  long long sA, long long sB, long long sC, long long sR,
                  int Mstore, int Nstore, int act)
{
    long long z = blockIdx.z;
    A += z*sA; B += z*sB;
    if (Cf) Cf += z*sC;
    if (Ch) Ch += z*sC;
    const float* Rp = resid ? resid + z*sR : nullptr;

    int wave = threadIdx.x >> 5;
    int lane = threadIdx.x & 31;
    int tm = blockIdx.y*8 + wave;
    if (tm >= Mtiles) return;           // wave-uniform exit, EXEC stays all-1 for WMMA
    int n0 = blockIdx.x*64;
    int lm = lane & 15;
    int hi = lane >> 4;

    // A fragment: lane<16 -> row M=lm holds K{0..15}; lane>=16 -> K{16..31}
    const __bf16* pa  = A + (long long)(tm*16 + lm)*lda + hi*16;
    // B fragment: lane = column n, low half-wave K{0..15}, high half-wave K{16..31}
    const __bf16* pb0 = B + (long long)(n0 +  0 + lm)*ldb + hi*16;
    const __bf16* pb1 = B + (long long)(n0 + 16 + lm)*ldb + hi*16;
    const __bf16* pb2 = B + (long long)(n0 + 32 + lm)*ldb + hi*16;
    const __bf16* pb3 = B + (long long)(n0 + 48 + lm)*ldb + hi*16;

    v8f acc0 = {}, acc1 = {}, acc2 = {}, acc3 = {};

    // software-pipelined K loop (branch-free; last iteration reloads same frags)
    v8bf al = *(const v8bf*)(pa);
    v8bf ah = *(const v8bf*)(pa + 8);
    v16bf a  = __builtin_shufflevector(al, ah, 0,1,2,3,4,5,6,7,8,9,10,11,12,13,14,15);
    v16bf b0 = *(const v16bf*)(pb0);
    v16bf b1 = *(const v16bf*)(pb1);
    v16bf b2 = *(const v16bf*)(pb2);
    v16bf b3 = *(const v16bf*)(pb3);

    for (int k0 = 0; k0 < K; k0 += 32) {
        int kn = (k0 + 32 < K) ? (k0 + 32) : k0;
        __builtin_prefetch(pa + kn + 32, 0, 0);
        v8bf nal = *(const v8bf*)(pa + kn);
        v8bf nah = *(const v8bf*)(pa + kn + 8);
        v16bf na  = __builtin_shufflevector(nal, nah, 0,1,2,3,4,5,6,7,8,9,10,11,12,13,14,15);
        v16bf nb0 = *(const v16bf*)(pb0 + kn);
        v16bf nb1 = *(const v16bf*)(pb1 + kn);
        v16bf nb2 = *(const v16bf*)(pb2 + kn);
        v16bf nb3 = *(const v16bf*)(pb3 + kn);

        acc0 = __builtin_amdgcn_wmma_f32_16x16x32_bf16(false, a, false, b0, (short)0, acc0, false, false);
        acc1 = __builtin_amdgcn_wmma_f32_16x16x32_bf16(false, a, false, b1, (short)0, acc1, false, false);
        acc2 = __builtin_amdgcn_wmma_f32_16x16x32_bf16(false, a, false, b2, (short)0, acc2, false, false);
        acc3 = __builtin_amdgcn_wmma_f32_16x16x32_bf16(false, a, false, b3, (short)0, acc3, false, false);

        a = na; b0 = nb0; b1 = nb1; b2 = nb2; b3 = nb3;
    }

    // C layout: VGPR r -> row (tm*16 + hi*8 + r), lane%16 -> column
    int mbase = tm*16 + hi*8;
    #pragma unroll
    for (int t = 0; t < 4; ++t) {
        v8f acc = (t==0) ? acc0 : (t==1) ? acc1 : (t==2) ? acc2 : acc3;
        int n = n0 + t*16 + lm;
        if (n >= Nstore) continue;
        float bv = bias ? bias[n] : 0.f;
        #pragma unroll
        for (int r = 0; r < 8; ++r) {
            int m = mbase + r;
            float v = acc[r] + bv;
            if (act == 1) v = gelu_f(v);
            else if (act == 2) v = softplus_f(v);
            if (m < Mstore) {
                if (Rp) v += Rp[(long long)m*ldcf + n];
                if (Cf) Cf[(long long)m*ldcf + n] = v;
                if (Ch) Ch[(long long)m*ldch + n] = (__bf16)v;
            }
        }
    }
}

// ---------------------------------------------------------------------------
// Elementwise / row kernels
// ---------------------------------------------------------------------------
__global__ void cvt_bf16_k(const float* __restrict__ s, __bf16* __restrict__ d, long long n){
    long long i = (long long)blockIdx.x*blockDim.x + threadIdx.x;
    if (i < n) d[i] = (__bf16)s[i];
}

__global__ void cvt_pad2d_k(const float* __restrict__ s, __bf16* __restrict__ d,
                            int Rs, int Rd, int Ks, int Kd){
    long long i = (long long)blockIdx.x*blockDim.x + threadIdx.x;
    if (i >= (long long)Rd*Kd) return;
    int k = (int)(i % Kd); int r = (int)(i / Kd);
    d[i] = (r < Rs && k < Ks) ? (__bf16)s[(long long)r*Ks + k] : (__bf16)0.f;
}

// One wave per row of 768; wave32 shfl reductions.
__global__ __launch_bounds__(256)
void layernorm_k(const float* __restrict__ x, long long inStride,
                 const float* __restrict__ w, const float* __restrict__ b,
                 __bf16* __restrict__ oh, long long ohStride,
                 float* __restrict__ of, long long ofStride, int rows)
{
    int wave = threadIdx.x >> 5, lane = threadIdx.x & 31;
    int row = blockIdx.x*8 + wave;
    if (row >= rows) return;
    const float* xr = x + (long long)row*inStride;
    float v[24]; float s = 0.f, ss = 0.f;
    #pragma unroll
    for (int i = 0; i < 24; ++i) { v[i] = xr[lane + i*32]; s += v[i]; ss += v[i]*v[i]; }
    #pragma unroll
    for (int o = 16; o; o >>= 1) { s += __shfl_xor(s, o, 32); ss += __shfl_xor(ss, o, 32); }
    float mean = s * (1.f/768.f);
    float var  = ss * (1.f/768.f) - mean*mean;
    float inv  = rsqrtf(var + 1e-5f);
    #pragma unroll
    for (int i = 0; i < 24; ++i) {
        int c = lane + i*32;
        float y = (v[i]-mean)*inv*w[c] + b[c];
        if (oh) oh[(long long)row*ohStride + c] = (__bf16)y;
        if (of) of[(long long)row*ofStride + c] = y;
    }
}

// Causal depthwise conv1d, k=4, + bias. Writes f32 and bf16.
__global__ void conv1d_k(const float* __restrict__ xn, const float* __restrict__ cw,
                         const float* __restrict__ cb,
                         float* __restrict__ xcf, __bf16* __restrict__ xch)
{
    long long i = (long long)blockIdx.x*blockDim.x + threadIdx.x;
    if (i >= (long long)MTOK*EMBED) return;
    int d = (int)(i % EMBED); long long bl = i / EMBED;
    int l = (int)(bl % SEQ); int b = (int)(bl / SEQ);
    float acc = cb[d];
    #pragma unroll
    for (int t = 0; t < KCONV; ++t) {
        int ls = l - (KCONV-1) + t;
        if (ls >= 0) acc += cw[d*KCONV + t] * xn[((long long)b*SEQ + ls)*EMBED + d];
    }
    xcf[i] = acc; xch[i] = (__bf16)acc;
}

__global__ void extract_dt_k(const float* __restrict__ xdbl, __bf16* __restrict__ dtA){
    long long i = (long long)blockIdx.x*blockDim.x + threadIdx.x;
    if (i >= (long long)MTOK*64) return;
    int k = (int)(i % 64); long long m = i/64;
    dtA[i] = (k < DTRK) ? (__bf16)xdbl[m*80 + k] : (__bf16)0.f;
}

// Selective scan: one thread per (b,d), 16 states in registers.
__global__ void ssm_scan_k(const float* __restrict__ dt, const float* __restrict__ xc,
                           const float* __restrict__ xdbl, const float* __restrict__ A_log,
                           const float* __restrict__ Dp, float* __restrict__ ys)
{
    int i = blockIdx.x*blockDim.x + threadIdx.x;
    if (i >= BATCH*EMBED) return;
    int d = i % EMBED, b = i / EMBED;
    float a[NSTATE], h[NSTATE];
    #pragma unroll
    for (int n = 0; n < NSTATE; ++n) { a[n] = -__expf(A_log[d*NSTATE+n]); h[n] = 0.f; }
    float dpv = Dp[d];
    for (int l = 0; l < SEQ; ++l) {
        long long m = (long long)b*SEQ + l;
        float dtv = dt[m*EMBED + d];
        float xcv = xc[m*EMBED + d];
        float dbx = dtv * xcv;
        const float* bc = xdbl + m*80;
        float y = 0.f;
        #pragma unroll
        for (int n = 0; n < NSTATE; ++n) {
            float dA = __expf(dtv * a[n]);
            h[n] = dA*h[n] + dbx*bc[48+n];
            y += h[n]*bc[64+n];
        }
        ys[m*EMBED + d] = y + xcv*dpv;
    }
}

// Pack QKV. Q: (BH, LP, 64); K: (BH, NKP, 64) zero-padded to 256 rows so the
// scores GEMM N-side needs no tile guards; V stored transposed (BH, 64, LP).
__global__ void qkv_pack_k(const float* __restrict__ qf, const float* __restrict__ kf,
                           const float* __restrict__ vf,
                           __bf16* __restrict__ qp, __bf16* __restrict__ kp,
                           __bf16* __restrict__ vt)
{
    long long i = (long long)blockIdx.x*blockDim.x + threadIdx.x;
    if (i >= (long long)BHH*NKP*DKH) return;
    int dk = (int)(i % DKH); long long r = i / DKH;
    int l = (int)(r % NKP); int bh = (int)(r / NKP);
    int b = bh / NH, h = bh % NH;
    float qv=0.f, kv=0.f, vv=0.f;
    if (l < SEQ) {
        long long src = ((long long)b*SEQ + l)*EMBED + h*DKH + dk;
        qv = qf[src]*0.125f;  // DK^-0.5
        kv = kf[src]; vv = vf[src];
    }
    kp[((long long)bh*NKP + l)*DKH + dk] = (__bf16)kv;
    if (l < LP) {
        qp[((long long)bh*LP + l)*DKH + dk] = (__bf16)qv;
        vt[((long long)bh*DKH + dk)*LP + l] = (__bf16)vv;
    }
}

// Softmax over 197 valid columns; zero the pad. One wave per row.
__global__ __launch_bounds__(256)
void softmax_k(const float* __restrict__ s, __bf16* __restrict__ p)
{
    int wave = threadIdx.x >> 5, lane = threadIdx.x & 31;
    long long row = (long long)blockIdx.x*8 + wave;
    if (row >= (long long)BHH*LP) return;
    int l = (int)(row % LP);
    const float* sr = s + row*LP;
    __bf16* pr = p + row*LP;
    if (l >= SEQ) {
        #pragma unroll
        for (int i = 0; i < 7; ++i) pr[lane + i*32] = (__bf16)0.f;
        return;
    }
    float vv[7]; float mx = -1e30f;
    #pragma unroll
    for (int i = 0; i < 7; ++i) {
        int c = lane + i*32;
        vv[i] = (c < SEQ) ? sr[c] : -1e30f;
        mx = fmaxf(mx, vv[i]);
    }
    #pragma unroll
    for (int o = 16; o; o >>= 1) mx = fmaxf(mx, __shfl_xor(mx, o, 32));
    float sum = 0.f;
    #pragma unroll
    for (int i = 0; i < 7; ++i) {
        int c = lane + i*32;
        vv[i] = (c < SEQ) ? __expf(vv[i]-mx) : 0.f;
        sum += vv[i];
    }
    #pragma unroll
    for (int o = 16; o; o >>= 1) sum += __shfl_xor(sum, o, 32);
    float inv = 1.f/sum;
    #pragma unroll
    for (int i = 0; i < 7; ++i) pr[lane + i*32] = (__bf16)(vv[i]*inv);
}

__global__ void attn_unpack_k(const float* __restrict__ of, __bf16* __restrict__ ao){
    long long i = (long long)blockIdx.x*blockDim.x + threadIdx.x;
    if (i >= (long long)MTOK*EMBED) return;
    int d = (int)(i % EMBED); long long bl = i/EMBED;
    int l = (int)(bl % SEQ); int b = (int)(bl / SEQ);
    int h = d / DKH, dk = d % DKH;
    ao[i] = (__bf16) of[(((long long)(b*NH+h))*LP + l)*DKH + dk];
}

// comb = tok + g*ys + (1-g)*attn_proj
__global__ void combine_k(const float* __restrict__ tok, const float* __restrict__ ys,
                          const float* __restrict__ ap, const float* __restrict__ gate,
                          int layer, float* __restrict__ comb)
{
    long long i = (long long)blockIdx.x*blockDim.x + threadIdx.x;
    if (i >= (long long)MTOK*EMBED) return;
    float g = gate[layer];
    comb[i] = tok[i] + g*ys[i] + (1.f-g)*ap[i];
}

// im2col for 16x16 stride-16 patches -> (3136, 768) bf16
__global__ void patch_gather_k(const float* __restrict__ x, __bf16* __restrict__ ap){
    long long i = (long long)blockIdx.x*blockDim.x + threadIdx.x;
    if (i >= (long long)BATCH*NPATCH*EMBED) return;
    int col = (int)(i % EMBED); long long row = i / EMBED;
    int p = (int)(row % NPATCH); int b = (int)(row / NPATCH);
    int ph = p / (IMG/PS), pw = p % (IMG/PS);
    int c = col / (PS*PS); int rem = col % (PS*PS); int ii = rem / PS, jj = rem % PS;
    ap[i] = (__bf16) x[(((long long)b*3 + c)*IMG + ph*PS + ii)*IMG + pw*PS + jj];
}

__global__ void assemble_tok_k(const float* __restrict__ cp, const float* __restrict__ pos,
                               const float* __restrict__ cls, float* __restrict__ tok){
    long long i = (long long)blockIdx.x*blockDim.x + threadIdx.x;
    if (i >= (long long)MTOK*EMBED) return;
    int d = (int)(i % EMBED); long long bl = i / EMBED;
    int l = (int)(bl % SEQ); int b = (int)(bl / SEQ);
    tok[i] = (l == 0) ? cls[d]
                      : cp[((long long)b*NPATCH + (l-1))*EMBED + d] + pos[(long long)(l-1)*EMBED + d];
}

// ---------------------------------------------------------------------------
// Host orchestration
// ---------------------------------------------------------------------------
extern "C" void kernel_launch(void* const* d_in, const int* in_sizes, int n_in,
                              void* d_out, int out_size, void* d_ws, size_t ws_size,
                              hipStream_t stream)
{
    (void)in_sizes; (void)n_in; (void)out_size; (void)ws_size;
    const float* X        = (const float*)d_in[0];
    const float* patch_w  = (const float*)d_in[1];
    const float* patch_b  = (const float*)d_in[2];
    const float* pos_emb  = (const float*)d_in[3];
    const float* cls_tok  = (const float*)d_in[4];
    const float* n1w = (const float*)d_in[5],  *n1b = (const float*)d_in[6];
    const float* n2w = (const float*)d_in[7],  *n2b = (const float*)d_in[8];
    const float* n3w = (const float*)d_in[9],  *n3b = (const float*)d_in[10];
    const float* A_log = (const float*)d_in[11];
    const float* Dp    = (const float*)d_in[12];
    const float* xproj_w  = (const float*)d_in[13];
    const float* dtproj_w = (const float*)d_in[14];
    const float* dtproj_b = (const float*)d_in[15];
    const float* conv_w   = (const float*)d_in[16];
    const float* conv_b   = (const float*)d_in[17];
    const float* wq = (const float*)d_in[18], *wk = (const float*)d_in[19], *wv = (const float*)d_in[20];
    const float* wo_w = (const float*)d_in[21], *wo_b = (const float*)d_in[22];
    const float* m1w = (const float*)d_in[23], *m1b = (const float*)d_in[24];
    const float* m2w = (const float*)d_in[25], *m2b = (const float*)d_in[26];
    const float* gate = (const float*)d_in[27];
    const float* fn_w = (const float*)d_in[28], *fn_b = (const float*)d_in[29];
    const float* head_w = (const float*)d_in[30], *head_b = (const float*)d_in[31];

    char* base = (char*)d_ws; size_t off = 0;
    auto carve = [&](size_t bytes)->void*{ void* p = base + off; off += (bytes + 255) & ~(size_t)255; return p; };

    const size_t S2 = (size_t)EMBED*EMBED;
    __bf16* w_q  = (__bf16*)carve(S2*2);
    __bf16* w_k  = (__bf16*)carve(S2*2);
    __bf16* w_v  = (__bf16*)carve(S2*2);
    __bf16* w_o  = (__bf16*)carve(S2*2);
    __bf16* w_xp = (__bf16*)carve((size_t)128*EMBED*2);   // 80 rows padded to 128
    __bf16* w_dt = (__bf16*)carve((size_t)EMBED*64*2);
    __bf16* w_m1 = (__bf16*)carve((size_t)MLPD*EMBED*2);
    __bf16* w_m2 = (__bf16*)carve((size_t)EMBED*MLPD*2);
    __bf16* w_pt = (__bf16*)carve(S2*2);
    __bf16* w_hd = (__bf16*)carve((size_t)1024*EMBED*2);  // 1000 rows padded to 1024

    __bf16* a_pt = (__bf16*)carve((size_t)BATCH*NPATCH*EMBED*2);
    float*  tok  = (float*)carve((size_t)MTOK*EMBED*4);
    float*  xn   = (float*)carve((size_t)MTOK*EMBED*4);
    float*  xcf  = (float*)carve((size_t)MTOK*EMBED*4);
    __bf16* xch  = (__bf16*)carve((size_t)MTOK*EMBED*2);
    float*  xdbl = (float*)carve((size_t)MTOK*80*4);
    __bf16* dtA  = (__bf16*)carve((size_t)MTOK*64*2);
    float*  dt_f = (float*)carve((size_t)MTOK*EMBED*4);
    float*  ys   = (float*)carve((size_t)MTOK*EMBED*4);
    __bf16* xa   = (__bf16*)carve((size_t)MTOK*EMBED*2);
    float*  qf   = (float*)carve((size_t)MTOK*EMBED*4);   // also reused for patch GEMM out
    float*  kf   = (float*)carve((size_t)MTOK*EMBED*4);
    float*  vf   = (float*)carve((size_t)MTOK*EMBED*4);
    __bf16* qp   = (__bf16*)carve((size_t)BHH*LP*DKH*2);
    __bf16* kp   = (__bf16*)carve((size_t)BHH*NKP*DKH*2);
    __bf16* vt   = (__bf16*)carve((size_t)BHH*DKH*LP*2);
    size_t scB = (size_t)BHH*LP*LP*4, hidB = (size_t)MTOK*MLPD*2;
    void* uni = carve(scB > hidB ? scB : hidB);           // scores / MLP-hidden union (disjoint lifetimes)
    float*  sc  = (float*)uni;
    __bf16* hid = (__bf16*)uni;
    __bf16* pr  = (__bf16*)carve((size_t)BHH*LP*LP*2);
    float*  ofb = (float*)carve((size_t)BHH*LP*DKH*4);
    __bf16* ao  = (__bf16*)carve((size_t)MTOK*EMBED*2);
    float*  apj = (float*)carve((size_t)MTOK*EMBED*4);
    float*  comb= (float*)carve((size_t)MTOK*EMBED*4);
    __bf16* h3  = (__bf16*)carve((size_t)MTOK*EMBED*2);
    __bf16* fts = (__bf16*)carve((size_t)BATCH*EMBED*2);

    auto g1 = [](long long n){ return dim3((unsigned)((n + 255)/256)); };
    // Nalloc must be a multiple of 64 (allocated/padded B rows).
    auto gemm = [&](const __bf16* A, const __bf16* B, float* Cf, __bf16* Ch,
                    const float* bias, const float* resid,
                    int M, int Nalloc, int K, int lda, int ldb, int ldcf, int ldch,
                    int batches, long long sA, long long sB, long long sC, long long sR,
                    int Mstore, int Nstore, int act){
        dim3 grid(Nalloc/64, (M/16 + 7)/8, batches);
        gemm_nt_bf16<<<grid, 256, 0, stream>>>(A,B,Cf,Ch,bias,resid, M/16, K,
                                               lda,ldb,ldcf,ldch, sA,sB,sC,sR, Mstore,Nstore,act);
    };

    // ---- Patch embedding: im2col + WMMA GEMM + token assembly ----
    cvt_bf16_k<<<g1(S2),256,0,stream>>>(patch_w, w_pt, (long long)S2);
    patch_gather_k<<<g1((long long)BATCH*NPATCH*EMBED),256,0,stream>>>(X, a_pt);
    gemm(a_pt, w_pt, qf, nullptr, patch_b, nullptr,
         BATCH*NPATCH, EMBED, EMBED, EMBED, EMBED, EMBED, 0,
         1, 0,0,0,0, BATCH*NPATCH, EMBED, 0);
    assemble_tok_k<<<g1((long long)MTOK*EMBED),256,0,stream>>>(qf, pos_emb, cls_tok, tok);

    for (int l = 0; l < DEPTH; ++l) {
        // per-layer weight conversion to bf16 (zero-padded where N/K need rounding)
        cvt_bf16_k<<<g1(S2),256,0,stream>>>(wq   + (size_t)l*S2, w_q, (long long)S2);
        cvt_bf16_k<<<g1(S2),256,0,stream>>>(wk   + (size_t)l*S2, w_k, (long long)S2);
        cvt_bf16_k<<<g1(S2),256,0,stream>>>(wv   + (size_t)l*S2, w_v, (long long)S2);
        cvt_bf16_k<<<g1(S2),256,0,stream>>>(wo_w + (size_t)l*S2, w_o, (long long)S2);
        cvt_pad2d_k<<<g1((long long)128*EMBED),256,0,stream>>>(xproj_w + (size_t)l*80*EMBED, w_xp, 80, 128, EMBED, EMBED);
        cvt_pad2d_k<<<g1((long long)EMBED*64),256,0,stream>>>(dtproj_w + (size_t)l*EMBED*DTRK, w_dt, EMBED, EMBED, DTRK, 64);
        cvt_bf16_k<<<g1((long long)MLPD*EMBED),256,0,stream>>>(m1w + (size_t)l*MLPD*EMBED, w_m1, (long long)MLPD*EMBED);
        cvt_bf16_k<<<g1((long long)EMBED*MLPD),256,0,stream>>>(m2w + (size_t)l*EMBED*MLPD, w_m2, (long long)EMBED*MLPD);

        // ---- SSM branch ----
        layernorm_k<<<(MTOK+7)/8,256,0,stream>>>(tok, EMBED, n1w + l*EMBED, n1b + l*EMBED,
                                                 (__bf16*)nullptr, 0, xn, EMBED, MTOK);
        conv1d_k<<<g1((long long)MTOK*EMBED),256,0,stream>>>(xn, conv_w + (size_t)l*EMBED*KCONV,
                                                             conv_b + l*EMBED, xcf, xch);
        gemm(xch, w_xp, xdbl, nullptr, nullptr, nullptr,
             MTOK, 128, EMBED, EMBED, EMBED, 80, 0, 1, 0,0,0,0, MTOK, 80, 0);
        extract_dt_k<<<g1((long long)MTOK*64),256,0,stream>>>(xdbl, dtA);
        gemm(dtA, w_dt, dt_f, nullptr, dtproj_b + l*EMBED, nullptr,
             MTOK, EMBED, 64, 64, 64, EMBED, 0, 1, 0,0,0,0, MTOK, EMBED, 2 /*softplus*/);
        ssm_scan_k<<<(BATCH*EMBED+255)/256,256,0,stream>>>(dt_f, xcf, xdbl,
                                                           A_log + (size_t)l*EMBED*NSTATE,
                                                           Dp + l*EMBED, ys);

        // ---- Attention branch ----
        layernorm_k<<<(MTOK+7)/8,256,0,stream>>>(tok, EMBED, n2w + l*EMBED, n2b + l*EMBED,
                                                 xa, EMBED, (float*)nullptr, 0, MTOK);
        gemm(xa, w_q, qf, nullptr, nullptr, nullptr, MTOK, EMBED, EMBED, EMBED,EMBED,EMBED,0, 1,0,0,0,0, MTOK,EMBED,0);
        gemm(xa, w_k, kf, nullptr, nullptr, nullptr, MTOK, EMBED, EMBED, EMBED,EMBED,EMBED,0, 1,0,0,0,0, MTOK,EMBED,0);
        gemm(xa, w_v, vf, nullptr, nullptr, nullptr, MTOK, EMBED, EMBED, EMBED,EMBED,EMBED,0, 1,0,0,0,0, MTOK,EMBED,0);
        qkv_pack_k<<<g1((long long)BHH*NKP*DKH),256,0,stream>>>(qf, kf, vf, qp, kp, vt);
        // S = (Q*scale) @ K^T  (192 batched 224x224x64 GEMMs; N padded to 256)
        gemm(qp, kp, sc, nullptr, nullptr, nullptr,
             LP, NKP, DKH, DKH, DKH, LP, 0,
             BHH, (long long)LP*DKH, (long long)NKP*DKH, (long long)LP*LP, 0, LP, LP, 0);
        softmax_k<<<((long long)BHH*LP+7)/8,256,0,stream>>>(sc, pr);
        // O = P @ V  (V stored transposed -> NT GEMM, K=224)
        gemm(pr, vt, ofb, nullptr, nullptr, nullptr,
             LP, DKH, LP, LP, LP, DKH, 0,
             BHH, (long long)LP*LP, (long long)DKH*LP, (long long)LP*DKH, 0, LP, DKH, 0);
        attn_unpack_k<<<g1((long long)MTOK*EMBED),256,0,stream>>>(ofb, ao);
        gemm(ao, w_o, apj, nullptr, wo_b + l*EMBED, nullptr,
             MTOK, EMBED, EMBED, EMBED,EMBED,EMBED,0, 1,0,0,0,0, MTOK,EMBED,0);

        // ---- Gate combine + MLP ----
        combine_k<<<g1((long long)MTOK*EMBED),256,0,stream>>>(tok, ys, apj, gate, l, comb);
        layernorm_k<<<(MTOK+7)/8,256,0,stream>>>(comb, EMBED, n3w + l*EMBED, n3b + l*EMBED,
                                                 h3, EMBED, (float*)nullptr, 0, MTOK);
        gemm(h3, w_m1, nullptr, hid, m1b + l*MLPD, nullptr,
             MTOK, MLPD, EMBED, EMBED, EMBED, 0, MLPD, 1, 0,0,0,0, MTOK, MLPD, 1 /*gelu*/);
        gemm(hid, w_m2, tok, nullptr, m2b + l*EMBED, comb,
             MTOK, EMBED, MLPD, MLPD, MLPD, EMBED, 0, 1, 0,0,0,0, MTOK, EMBED, 0);
    }

    // ---- Final LN (cls rows only) + classification head ----
    layernorm_k<<<(BATCH+7)/8,256,0,stream>>>(tok, (long long)SEQ*EMBED, fn_w, fn_b,
                                              fts, EMBED, (float*)nullptr, 0, BATCH);
    cvt_pad2d_k<<<g1((long long)1024*EMBED),256,0,stream>>>(head_w, w_hd, NCLS, 1024, EMBED, EMBED);
    gemm(fts, w_hd, (float*)d_out, nullptr, head_b, nullptr,
         BATCH, 1024, EMBED, EMBED, EMBED, NCLS, 0, 1, 0,0,0,0, BATCH, NCLS, 0);
}
